// MultiHeadAttention_86715389706697
// MI455X (gfx1250) — matrix operand order repaired and checked by reading, not compiled, not run on hardware
//
#include <hip/hip_runtime.h>
#include <hip/hip_bf16.h>

// ---------------------------------------------------------------------------
// Types for CDNA5 WMMA (wave32): v_wmma_f32_16x16x32_bf16
// ---------------------------------------------------------------------------
typedef __attribute__((ext_vector_type(16))) __bf16 bf16x16;
typedef __attribute__((ext_vector_type(8)))  __bf16 bf16x8;
typedef __attribute__((ext_vector_type(8)))  float  f32x8;
typedef __attribute__((ext_vector_type(4)))  int    i32x4;

static constexpr int BATCH = 4;
static constexpr int HEADS = 12;
static constexpr int SEQ   = 1024;
static constexpr int DM    = 768;
static constexpr int DK    = 64;
static constexpr int BH    = BATCH * HEADS;   // 48
static constexpr int ROWS  = BATCH * SEQ;     // 4096

__device__ __forceinline__ f32x8 zero8() {
    f32x8 z;
#pragma unroll
    for (int i = 0; i < 8; ++i) z[i] = 0.0f;
    return z;
}

__device__ __forceinline__ bf16x16 join16(bf16x8 lo, bf16x8 hi) {
    bf16x16 r;
#pragma unroll
    for (int i = 0; i < 8; ++i) { r[i] = lo[i]; r[8 + i] = hi[i]; }
    return r;
}

// A fragment 16x32 bf16 (ISA 7.12.2): lanes 0-15 rows M=0..15 hold K [0,8)+[16,24),
// lanes 16-31 same rows hold K [8,16)+[24,32). Two 16-byte LDS loads.
__device__ __forceinline__ bf16x16 load_a_frag(const __bf16* lds, int stride, int lane) {
    const int m  = lane & 15;
    const int kb = (lane >> 4) * 8;
    const __bf16* p = lds + m * stride + kb;
    bf16x8 lo = *(const bf16x8*)(p);
    bf16x8 hi = *(const bf16x8*)(p + 16);
    return join16(lo, hi);
}

// B fragment 32x16 bf16: lane (L&15)=column N, lanes 0-15 hold K [0,8)+[16,24), lanes 16-31 K [8,16)+[24,32).
// LDS tile must be N-major: lds[n][k], row stride = `stride` elements.
__device__ __forceinline__ bf16x16 load_b_frag(const __bf16* lds, int stride, int lane) {
    const int n  = lane & 15;
    const int kb = (lane >> 4) * 8;
    const __bf16* p = lds + n * stride + kb;
    bf16x8 lo = *(const bf16x8*)(p);
    bf16x8 hi = *(const bf16x8*)(p + 16);
    return join16(lo, hi);
}

__device__ __forceinline__ f32x8 wmma_bf16(bf16x16 a, bf16x16 b, f32x8 c) {
    return __builtin_amdgcn_wmma_f32_16x16x32_bf16(false, a, false, b, (short)0, c,
                                                   false, false);
}

// ---------------------------------------------------------------------------
// Async LDS-DMA copy path (GLOBAL_LOAD_ASYNC_TO_LDS_B128, tracked on ASYNCcnt).
// Probe result from the previous round: the builtin exists and its pointer
// params are i32x4* in AS(1)/AS(3).  Fallback is the VGPR-staged copy.
// ---------------------------------------------------------------------------
#if defined(__has_builtin)
#  if __has_builtin(__builtin_amdgcn_global_load_async_to_lds_b128)
#    define HAVE_ASYNC_LDS 1
#  endif
#endif
#ifndef HAVE_ASYNC_LDS
#  define HAVE_ASYNC_LDS 0
#endif

__device__ __forceinline__ void async_copy_b128(const void* g, void* l) {
#if HAVE_ASYNC_LDS
    __builtin_amdgcn_global_load_async_to_lds_b128(
        (__attribute__((address_space(1))) i32x4*)g,
        (__attribute__((address_space(3))) i32x4*)l, 0, 0);
#else
    *(bf16x8*)l = *(const bf16x8*)g;
#endif
}

__device__ __forceinline__ void async_join() {
#if HAVE_ASYNC_LDS
#  if defined(__has_builtin)
#    if __has_builtin(__builtin_amdgcn_s_wait_asynccnt)
    __builtin_amdgcn_s_wait_asynccnt(0);
#    endif
#  endif
#endif
}

// ---------------------------------------------------------------------------
// K0: fp32 -> bf16 weight conversion
// ---------------------------------------------------------------------------
__global__ void k_cvt_bf16(const float* __restrict__ src, __bf16* __restrict__ dst, int n) {
    int i = blockIdx.x * blockDim.x + threadIdx.x;
    if (i < n) dst[i] = (__bf16)src[i];
}

// ---------------------------------------------------------------------------
// K1: projection GEMM  X[4096x768] (fp32) @ W[768x768] (bf16) -> head-split
//     Out[b][h][s][d] bf16.  Block tile 128x128, 8 waves (4x2), wave 32x64.
// ---------------------------------------------------------------------------
__global__ __launch_bounds__(256) void k_proj(const float* __restrict__ X,
                                              const __bf16* __restrict__ W,
                                              __bf16* __restrict__ Out) {
    __shared__ __align__(16) __bf16 As[128][32];   // row-major [m][k]
    __shared__ __align__(16) __bf16 Bs[128][32];   // N-major  [n][k]
    const int tid  = threadIdx.x;
    const int lane = tid & 31;
    const int wid  = tid >> 5;
    const int m0   = (int)blockIdx.x * 128;
    const int n0   = (int)blockIdx.y * 128;
    const int mOff = (wid >> 1) * 32;
    const int nOff = (wid & 1) * 64;

    f32x8 acc[2][4];
#pragma unroll
    for (int m = 0; m < 2; ++m)
#pragma unroll
        for (int n = 0; n < 4; ++n) acc[m][n] = zero8();

    for (int kk = 0; kk < DM; kk += 32) {
        {   // A tile: 128x32 fp32 -> bf16 (conversion -> must go through VGPRs)
            const int row = tid >> 1;
            const int kc  = (tid & 1) * 16;
            const float* src = X + (size_t)(m0 + row) * DM + kk + kc;
#pragma unroll
            for (int j = 0; j < 16; ++j) As[row][kc + j] = (__bf16)src[j];
        }
        {   // B tile: 32x128 bf16, transpose into N-major LDS
            const int k_ = tid >> 3;
            const int nc = (tid & 7) * 16;
            const __bf16* src = W + (size_t)(kk + k_) * DM + n0 + nc;
#pragma unroll
            for (int j = 0; j < 16; ++j) Bs[nc + j][k_] = src[j];
        }
        __syncthreads();
        bf16x16 bfr[4];
#pragma unroll
        for (int n = 0; n < 4; ++n) bfr[n] = load_b_frag(&Bs[nOff + n * 16][0], 32, lane);
#pragma unroll
        for (int m = 0; m < 2; ++m) {
            bf16x16 afr = load_a_frag(&As[mOff + m * 16][0], 32, lane);
#pragma unroll
            for (int n = 0; n < 4; ++n) acc[m][n] = wmma_bf16(afr, bfr[n], acc[m][n]);
        }
        __syncthreads();
    }
    // epilogue: C/D layout -> head-split bf16
#pragma unroll
    for (int m = 0; m < 2; ++m) {
        const int rbase = m0 + mOff + m * 16 + (lane >> 4) * 8;
#pragma unroll
        for (int n = 0; n < 4; ++n) {
            const int c = n0 + nOff + n * 16 + (lane & 15);
            const int h = c >> 6, d = c & 63;
#pragma unroll
            for (int i = 0; i < 8; ++i) {
                const int r = rbase + i;
                const int b = r >> 10, s = r & 1023;
                Out[(((size_t)(b * HEADS + h)) * SEQ + s) * DK + d] = (__bf16)acc[m][n][i];
            }
        }
    }
}

// ---------------------------------------------------------------------------
// K2: scores[b,h,s,t] = (Q . K) / 8, masked by mask1_2[b,h,t,s].
//     NT GEMM: both Q and K tiles are d-contiguous -> no transpose; tiles are
//     staged into LDS with the async LDS-DMA path (ASYNCcnt).
// ---------------------------------------------------------------------------
__global__ __launch_bounds__(256) void k_scores(const __bf16* __restrict__ Qb,
                                                const __bf16* __restrict__ Kb,
                                                const unsigned char* __restrict__ Mask,
                                                float* __restrict__ Sc) {
    __shared__ __align__(16) __bf16 Qs[128][64];
    __shared__ __align__(16) __bf16 Ks[128][64];
    const int tid  = threadIdx.x;
    const int lane = tid & 31;
    const int wid  = tid >> 5;
    const int bh   = blockIdx.z;
    const int s0   = (int)blockIdx.x * 128;
    const int t0   = (int)blockIdx.y * 128;
    const __bf16* Qp = Qb + (size_t)bh * SEQ * DK;
    const __bf16* Kp = Kb + (size_t)bh * SEQ * DK;
    {
        const int row = tid >> 1;
        const int c0  = (tid & 1) * 32;
        const __bf16* q = Qp + (size_t)(s0 + row) * DK + c0;
        const __bf16* k = Kp + (size_t)(t0 + row) * DK + c0;
#pragma unroll
        for (int j = 0; j < 4; ++j) async_copy_b128(q + j * 8, &Qs[row][c0 + j * 8]);
#pragma unroll
        for (int j = 0; j < 4; ++j) async_copy_b128(k + j * 8, &Ks[row][c0 + j * 8]);
    }
    async_join();
    __syncthreads();

    const int sOff = (wid >> 1) * 32;
    const int tOff = (wid & 1) * 64;
    f32x8 acc[2][4];
#pragma unroll
    for (int m = 0; m < 2; ++m)
#pragma unroll
        for (int n = 0; n < 4; ++n) acc[m][n] = zero8();

#pragma unroll
    for (int kk = 0; kk < DK; kk += 32) {
        bf16x16 bfr[4];
#pragma unroll
        for (int n = 0; n < 4; ++n) bfr[n] = load_b_frag(&Ks[tOff + n * 16][kk], 64, lane);
#pragma unroll
        for (int m = 0; m < 2; ++m) {
            bf16x16 afr = load_a_frag(&Qs[sOff + m * 16][kk], 64, lane);
#pragma unroll
            for (int n = 0; n < 4; ++n) acc[m][n] = wmma_bf16(afr, bfr[n], acc[m][n]);
        }
    }

    const unsigned char* Mp = Mask + (size_t)bh * SEQ * SEQ;
    float* Sp = Sc + (size_t)bh * SEQ * SEQ;
#pragma unroll
    for (int m = 0; m < 2; ++m) {
        const int sb = s0 + sOff + m * 16 + (lane >> 4) * 8;
#pragma unroll
        for (int n = 0; n < 4; ++n) {
            const int t = t0 + tOff + n * 16 + (lane & 15);
#pragma unroll
            for (int i = 0; i < 8; ++i) {
                const int s = sb + i;
                float v = acc[m][n][i] * 0.125f;          // 1/sqrt(64)
                if (Mp[(size_t)t * SEQ + s]) v = -1e9f;   // mask1_2.transpose(2,3)
                Sp[(size_t)s * SEQ + t] = v;
            }
        }
    }
}

// ---------------------------------------------------------------------------
// K3: per-row softmax stats (max, sum of exp) over t
// ---------------------------------------------------------------------------
__global__ __launch_bounds__(256) void k_rowstats(const float* __restrict__ Sc,
                                                  float* __restrict__ rmax,
                                                  float* __restrict__ rsum) {
    __shared__ float red[256];
    const int row = blockIdx.x;                 // bh*1024 + s
    const float* p = Sc + (size_t)row * SEQ;
    float v[4];
#pragma unroll
    for (int j = 0; j < 4; ++j) v[j] = p[threadIdx.x + j * 256];
    float m = fmaxf(fmaxf(v[0], v[1]), fmaxf(v[2], v[3]));
    red[threadIdx.x] = m;
    __syncthreads();
    for (int off = 128; off > 0; off >>= 1) {
        if (threadIdx.x < off) red[threadIdx.x] = fmaxf(red[threadIdx.x], red[threadIdx.x + off]);
        __syncthreads();
    }
    m = red[0];
    __syncthreads();
    float s = 0.0f;
#pragma unroll
    for (int j = 0; j < 4; ++j) s += __expf(v[j] - m);
    red[threadIdx.x] = s;
    __syncthreads();
    for (int off = 128; off > 0; off >>= 1) {
        if (threadIdx.x < off) red[threadIdx.x] += red[threadIdx.x + off];
        __syncthreads();
    }
    if (threadIdx.x == 0) { rmax[row] = m; rsum[row] = red[0]; }
}

// ---------------------------------------------------------------------------
// K4: per-column online-softmax stats over s (coalesced: tid -> column).
//     4-way unrolled so the fmax/exp of 4 rows issue independently before the
//     serial running-state merge.
// ---------------------------------------------------------------------------
__global__ __launch_bounds__(128) void k_colstats(const float* __restrict__ Sc,
                                                  float* __restrict__ cmax,
                                                  float* __restrict__ csum) {
    const int bh = blockIdx.y;
    const int t  = (int)blockIdx.x * 128 + threadIdx.x;
    const float* p = Sc + (size_t)bh * SEQ * SEQ + t;
    float m = -3.0e38f, s = 0.0f;
    for (int srow = 0; srow < SEQ; srow += 4) {
        if (srow + 32 < SEQ) __builtin_prefetch(p + (size_t)(srow + 32) * SEQ, 0, 1);
        float v0 = p[(size_t)(srow + 0) * SEQ];
        float v1 = p[(size_t)(srow + 1) * SEQ];
        float v2 = p[(size_t)(srow + 2) * SEQ];
        float v3 = p[(size_t)(srow + 3) * SEQ];
        const float lm = fmaxf(fmaxf(v0, v1), fmaxf(v2, v3));
        const float ls = __expf(v0 - lm) + __expf(v1 - lm) +
                         __expf(v2 - lm) + __expf(v3 - lm);
        const float nm = fmaxf(m, lm);
        s = s * __expf(m - nm) + ls * __expf(lm - nm);
        m = nm;
    }
    cmax[bh * SEQ + t] = m;
    csum[bh * SEQ + t] = s;
}

// ---------------------------------------------------------------------------
// K5: ctx1[s,d] = sum_t attn1[s,t] * K[t,d]; softmax fused into A-tile loader.
//     Block 128(s) x 64(d); wave 32x32 (2x2 frags); K tile transposed to d-major.
// ---------------------------------------------------------------------------
__global__ __launch_bounds__(256) void k_ctx1(const float* __restrict__ Sc,
                                              const float* __restrict__ rmax,
                                              const float* __restrict__ rsum,
                                              const __bf16* __restrict__ Kb,
                                              __bf16* __restrict__ C1) {
    __shared__ __align__(16) __bf16 As[128][32];   // [s][t-rel]
    __shared__ __align__(16) __bf16 Bs[64][32];    // d-major: [d][t-rel]
    const int tid  = threadIdx.x;
    const int lane = tid & 31;
    const int wid  = tid >> 5;
    const int bh   = blockIdx.y;
    const int b = bh / HEADS, h = bh % HEADS;
    const int s0 = (int)blockIdx.x * 128;
    const float* Sp = Sc + (size_t)bh * SEQ * SEQ;
    const __bf16* Kp = Kb + (size_t)bh * SEQ * DK;
    const int sOff = (wid >> 1) * 32;
    const int dOff = (wid & 1) * 32;

    f32x8 acc[2][2];
#pragma unroll
    for (int m = 0; m < 2; ++m)
#pragma unroll
        for (int n = 0; n < 2; ++n) acc[m][n] = zero8();

    const int arow = tid >> 1;
    const int akc  = (tid & 1) * 16;
    const float rm = rmax[bh * SEQ + s0 + arow];
    const float ri = 1.0f / rsum[bh * SEQ + s0 + arow];

    for (int kk = 0; kk < SEQ; kk += 32) {
        {   // A tile: normalize scores row -> attn1 bf16
            const float* src = Sp + (size_t)(s0 + arow) * SEQ + kk + akc;
#pragma unroll
            for (int j = 0; j < 16; ++j)
                As[arow][akc + j] = (__bf16)(__expf(src[j] - rm) * ri);
        }
        {   // B tile: K[t][d] (32x64) -> Bs[d][t]
            const int trow = tid >> 3;
            const int dc   = (tid & 7) * 8;
            const __bf16* src = Kp + (size_t)(kk + trow) * DK + dc;
#pragma unroll
            for (int j = 0; j < 8; ++j) Bs[dc + j][trow] = src[j];
        }
        __syncthreads();
        bf16x16 bfr[2];
#pragma unroll
        for (int n = 0; n < 2; ++n) bfr[n] = load_b_frag(&Bs[dOff + n * 16][0], 32, lane);
#pragma unroll
        for (int m = 0; m < 2; ++m) {
            bf16x16 afr = load_a_frag(&As[sOff + m * 16][0], 32, lane);
#pragma unroll
            for (int n = 0; n < 2; ++n) acc[m][n] = wmma_bf16(afr, bfr[n], acc[m][n]);
        }
        __syncthreads();
    }
#pragma unroll
    for (int m = 0; m < 2; ++m) {
        const int sb = s0 + sOff + m * 16 + (lane >> 4) * 8;
#pragma unroll
        for (int n = 0; n < 2; ++n) {
            const int d = dOff + n * 16 + (lane & 15);
#pragma unroll
            for (int i = 0; i < 8; ++i) {
                const int s = sb + i;
                C1[((size_t)(b * SEQ + s)) * DM + h * DK + d] = (__bf16)acc[m][n][i];
            }
        }
    }
}

// ---------------------------------------------------------------------------
// K6: ctx2[t,d] = sum_s attn2[t,s] * Q[s,d]; column-softmax fused, scores
//     transposed during LDS staging (coalesced global reads).
// ---------------------------------------------------------------------------
__global__ __launch_bounds__(256) void k_ctx2(const float* __restrict__ Sc,
                                              const float* __restrict__ cmax,
                                              const float* __restrict__ csum,
                                              const __bf16* __restrict__ Qb,
                                              __bf16* __restrict__ C2) {
    __shared__ __align__(16) __bf16 As[128][32];   // [t][s-rel]
    __shared__ __align__(16) __bf16 Bs[64][32];    // d-major: [d][s-rel]
    const int tid  = threadIdx.x;
    const int lane = tid & 31;
    const int wid  = tid >> 5;
    const int bh   = blockIdx.y;
    const int b = bh / HEADS, h = bh % HEADS;
    const int t0 = (int)blockIdx.x * 128;
    const float* Sp = Sc + (size_t)bh * SEQ * SEQ;
    const __bf16* Qp = Qb + (size_t)bh * SEQ * DK;
    const int tOff = (wid >> 1) * 32;
    const int dOff = (wid & 1) * 32;

    f32x8 acc[2][2];
#pragma unroll
    for (int m = 0; m < 2; ++m)
#pragma unroll
        for (int n = 0; n < 2; ++n) acc[m][n] = zero8();

    // A-loader thread geometry: fixed columns -> hoist column stats
    const int asi = tid >> 3;            // s-rel row of scores, 0..31
    const int atc = (tid & 7) * 16;      // 16 columns per thread
    float cm[16], ci[16];
#pragma unroll
    for (int j = 0; j < 16; ++j) {
        const int tg = t0 + atc + j;
        cm[j] = cmax[bh * SEQ + tg];
        ci[j] = 1.0f / csum[bh * SEQ + tg];
    }

    for (int kk = 0; kk < SEQ; kk += 32) {
        {   // A tile: read scores[s][t] rows, normalize per column, store transposed
            const float* src = Sp + (size_t)(kk + asi) * SEQ + t0 + atc;
#pragma unroll
            for (int j = 0; j < 16; ++j)
                As[atc + j][asi] = (__bf16)(__expf(src[j] - cm[j]) * ci[j]);
        }
        {   // B tile: Q[s][d] (32x64) -> Bs[d][s]
            const int srow = tid >> 3;
            const int dc   = (tid & 7) * 8;
            const __bf16* src = Qp + (size_t)(kk + srow) * DK + dc;
#pragma unroll
            for (int j = 0; j < 8; ++j) Bs[dc + j][srow] = src[j];
        }
        __syncthreads();
        bf16x16 bfr[2];
#pragma unroll
        for (int n = 0; n < 2; ++n) bfr[n] = load_b_frag(&Bs[dOff + n * 16][0], 32, lane);
#pragma unroll
        for (int m = 0; m < 2; ++m) {
            bf16x16 afr = load_a_frag(&As[tOff + m * 16][0], 32, lane);
#pragma unroll
            for (int n = 0; n < 2; ++n) acc[m][n] = wmma_bf16(afr, bfr[n], acc[m][n]);
        }
        __syncthreads();
    }
#pragma unroll
    for (int m = 0; m < 2; ++m) {
        const int tb = t0 + tOff + m * 16 + (lane >> 4) * 8;
#pragma unroll
        for (int n = 0; n < 2; ++n) {
            const int d = dOff + n * 16 + (lane & 15);
#pragma unroll
            for (int i = 0; i < 8; ++i) {
                const int t = tb + i;
                C2[((size_t)(b * SEQ + t)) * DM + h * DK + d] = (__bf16)acc[m][n][i];
            }
        }
    }
}

// ---------------------------------------------------------------------------
// K7: fc GEMM  C[4096x768] (bf16) @ fc1[768x768] (bf16) -> Y fp32.
//     A tile is a straight bf16 copy -> async LDS-DMA path.
// ---------------------------------------------------------------------------
__global__ __launch_bounds__(256) void k_fc(const __bf16* __restrict__ Cin,
                                            const __bf16* __restrict__ W,
                                            float* __restrict__ Y) {
    __shared__ __align__(16) __bf16 As[128][32];
    __shared__ __align__(16) __bf16 Bs[128][32];
    const int tid  = threadIdx.x;
    const int lane = tid & 31;
    const int wid  = tid >> 5;
    const int m0   = (int)blockIdx.x * 128;
    const int n0   = (int)blockIdx.y * 128;
    const int mOff = (wid >> 1) * 32;
    const int nOff = (wid & 1) * 64;

    f32x8 acc[2][4];
#pragma unroll
    for (int m = 0; m < 2; ++m)
#pragma unroll
        for (int n = 0; n < 4; ++n) acc[m][n] = zero8();

    for (int kk = 0; kk < DM; kk += 32) {
        {   // A tile: direct bf16 copy via async LDS-DMA
            const int row = tid >> 1;
            const int kc  = (tid & 1) * 16;
            const __bf16* src = Cin + (size_t)(m0 + row) * DM + kk + kc;
            async_copy_b128(src,     &As[row][kc]);
            async_copy_b128(src + 8, &As[row][kc + 8]);
        }
        {   // B tile: transpose into N-major LDS
            const int k_ = tid >> 3;
            const int nc = (tid & 7) * 16;
            const __bf16* src = W + (size_t)(kk + k_) * DM + n0 + nc;
#pragma unroll
            for (int j = 0; j < 16; ++j) Bs[nc + j][k_] = src[j];
        }
        async_join();
        __syncthreads();
        bf16x16 bfr[4];
#pragma unroll
        for (int n = 0; n < 4; ++n) bfr[n] = load_b_frag(&Bs[nOff + n * 16][0], 32, lane);
#pragma unroll
        for (int m = 0; m < 2; ++m) {
            bf16x16 afr = load_a_frag(&As[mOff + m * 16][0], 32, lane);
#pragma unroll
            for (int n = 0; n < 4; ++n) acc[m][n] = wmma_bf16(afr, bfr[n], acc[m][n]);
        }
        __syncthreads();
    }
#pragma unroll
    for (int m = 0; m < 2; ++m) {
        const int rb = m0 + mOff + m * 16 + (lane >> 4) * 8;
#pragma unroll
        for (int n = 0; n < 4; ++n) {
            const int c = n0 + nOff + n * 16 + (lane & 15);
#pragma unroll
            for (int i = 0; i < 8; ++i)
                Y[(size_t)(rb + i) * DM + c] = acc[m][n][i];
        }
    }
}

// ---------------------------------------------------------------------------
// K8: LayerNorm over rows of 768, write fp32 output
// ---------------------------------------------------------------------------
__global__ __launch_bounds__(256) void k_ln(const float* __restrict__ Y,
                                            const float* __restrict__ g,
                                            const float* __restrict__ beta,
                                            float* __restrict__ Out) {
    __shared__ float red[256];
    const int row = blockIdx.x;
    const float* p = Y + (size_t)row * DM;
    float v[3];
#pragma unroll
    for (int j = 0; j < 3; ++j) v[j] = p[threadIdx.x + j * 256];
    red[threadIdx.x] = v[0] + v[1] + v[2];
    __syncthreads();
    for (int off = 128; off > 0; off >>= 1) {
        if (threadIdx.x < off) red[threadIdx.x] += red[threadIdx.x + off];
        __syncthreads();
    }
    const float mean = red[0] * (1.0f / DM);
    __syncthreads();
    float d2 = 0.0f;
#pragma unroll
    for (int j = 0; j < 3; ++j) { const float d = v[j] - mean; d2 += d * d; }
    red[threadIdx.x] = d2;
    __syncthreads();
    for (int off = 128; off > 0; off >>= 1) {
        if (threadIdx.x < off) red[threadIdx.x] += red[threadIdx.x + off];
        __syncthreads();
    }
    const float inv = rsqrtf(red[0] * (1.0f / DM) + 1e-5f);
#pragma unroll
    for (int j = 0; j < 3; ++j) {
        const int c = threadIdx.x + j * 256;
        Out[(size_t)row * DM + c] = (v[j] - mean) * inv * g[c] + beta[c];
    }
}

// ---------------------------------------------------------------------------
// Host launcher
// ---------------------------------------------------------------------------
extern "C" void kernel_launch(void* const* d_in, const int* in_sizes, int n_in,
                              void* d_out, int out_size, void* d_ws, size_t ws_size,
                              hipStream_t stream) {
    (void)in_sizes; (void)n_in; (void)out_size; (void)ws_size;
    const float* pro1 = (const float*)d_in[0];
    const float* pro2 = (const float*)d_in[1];
    const unsigned char* mask = (const unsigned char*)d_in[2];
    const float* W_Q  = (const float*)d_in[3];
    const float* W_K  = (const float*)d_in[4];
    const float* fc1  = (const float*)d_in[5];
    const float* g1   = (const float*)d_in[6];
    const float* b1   = (const float*)d_in[7];
    const float* g2   = (const float*)d_in[8];
    const float* b2   = (const float*)d_in[9];
    float* out = (float*)d_out;

    char* ws = (char*)d_ws;
    size_t off = 0;
    auto take = [&](size_t bytes) -> char* {
        char* p = ws + off;
        off = (off + bytes + 255) & ~(size_t)255;
        return p;
    };
    const size_t SC_B  = (size_t)BH * SEQ * SEQ * sizeof(float);      // 192 MB
    const size_t QK_B  = (size_t)BH * SEQ * DK * sizeof(__bf16);      // 6 MB
    const size_t W_B   = (size_t)DM * DM * sizeof(__bf16);
    const size_t ST_B  = (size_t)BH * SEQ * sizeof(float);
    const size_t C_B   = (size_t)ROWS * DM * sizeof(__bf16);
    const size_t Y_B   = (size_t)ROWS * DM * sizeof(float);

    float*  Sc   = (float*)take(SC_B);
    __bf16* Qb   = (__bf16*)take(QK_B);
    __bf16* Kb   = (__bf16*)take(QK_B);
    __bf16* Wqb  = (__bf16*)take(W_B);
    __bf16* Wkb  = (__bf16*)take(W_B);
    __bf16* Fcb  = (__bf16*)take(W_B);
    float*  rmax = (float*)take(ST_B);
    float*  rsum = (float*)take(ST_B);
    float*  cmax = (float*)take(ST_B);
    float*  csum = (float*)take(ST_B);
    __bf16* C1   = (__bf16*)take(C_B);
    __bf16* C2   = (__bf16*)take(C_B);
    float*  Y1   = (float*)take(Y_B);
    float*  Y2   = (float*)take(Y_B);

    const int nW = DM * DM;
    k_cvt_bf16<<<(nW + 255) / 256, 256, 0, stream>>>(W_Q, Wqb, nW);
    k_cvt_bf16<<<(nW + 255) / 256, 256, 0, stream>>>(W_K, Wkb, nW);
    k_cvt_bf16<<<(nW + 255) / 256, 256, 0, stream>>>(fc1, Fcb, nW);

    dim3 gproj(ROWS / 128, DM / 128);
    k_proj<<<gproj, 256, 0, stream>>>(pro1, Wqb, Qb);
    k_proj<<<gproj, 256, 0, stream>>>(pro2, Wkb, Kb);

    dim3 gsc(SEQ / 128, SEQ / 128, BH);
    k_scores<<<gsc, 256, 0, stream>>>(Qb, Kb, mask, Sc);

    k_rowstats<<<BH * SEQ, 256, 0, stream>>>(Sc, rmax, rsum);
    dim3 gcol(SEQ / 128, BH);
    k_colstats<<<gcol, 128, 0, stream>>>(Sc, cmax, csum);

    dim3 gctx(SEQ / 128, BH);
    k_ctx1<<<gctx, 256, 0, stream>>>(Sc, rmax, rsum, Kb, C1);
    k_ctx2<<<gctx, 256, 0, stream>>>(Sc, cmax, csum, Qb, C2);

    dim3 gfc(ROWS / 128, DM / 128);
    k_fc<<<gfc, 256, 0, stream>>>(C1, Fcb, Y1);
    k_fc<<<gfc, 256, 0, stream>>>(C2, Fcb, Y2);

    k_ln<<<ROWS, 256, 0, stream>>>(Y1, g1, b1, out);
    k_ln<<<ROWS, 256, 0, stream>>>(Y2, g2, b2, out + (size_t)ROWS * DM);
}
